// BertSelfAttention_54030688584172
// MI455X (gfx1250) — matrix operand order repaired and checked by reading, not compiled, not run on hardware
//
#include <hip/hip_runtime.h>
#include <hip/hip_bf16.h>

// ---------------------------------------------------------------------------
// BERT self-attention for MI455X (gfx1250), wave32 + WMMA bf16 + async-LDS.
//   B=8, L=1024, H=1024, NH=16, HD=64, scale = 1/8.
// Kernel 1: fused QKV projection GEMM (fp32 -> bf16 WMMA -> bf16 workspace)
// Kernel 2: flash attention over bf16 Q/K/V, fp32 output.
// ---------------------------------------------------------------------------

typedef __attribute__((ext_vector_type(16))) __bf16 v16bf;
typedef __attribute__((ext_vector_type(8)))  float  v8f;

union Frag16 { v16bf v; uint4 q[2]; };

#define WMMA_BF16(A, Bm, C) \
  __builtin_amdgcn_wmma_f32_16x16x32_bf16(false, (A), false, (Bm), (short)0, (C), false, false)

__device__ __forceinline__ unsigned short f32_to_bf16_rne(float f) {
  union { float f; unsigned int u; } x; x.f = f;
  unsigned int u = x.u + (0x7FFFu + ((x.u >> 16) & 1u));
  return (unsigned short)(u >> 16);
}

__device__ __forceinline__ unsigned int pack2_bf16(float a, float b) {
  return (unsigned int)f32_to_bf16_rne(a) | ((unsigned int)f32_to_bf16_rne(b) << 16);
}

// ---------------------------------------------------------------------------
// Kernel 1: Y = X @ W^T + bias for W in {Wq,Wk,Wv}.
// X: [8192,1024] fp32 row-major.  W: [1024,1024] fp32 row-major ([out,in]).
// Output: bf16 in [B, nh, L, 64] layout (per projection workspace).
// Block: 256 threads (8 waves), tile 128(M) x 128(N); wave tile 64x32.
// ---------------------------------------------------------------------------
__global__ __launch_bounds__(256)
void qkv_proj_kernel(const float* __restrict__ hs,
                     const float* __restrict__ Wq, const float* __restrict__ bq,
                     const float* __restrict__ Wk, const float* __restrict__ bk,
                     const float* __restrict__ Wv, const float* __restrict__ bv,
                     unsigned short* __restrict__ qws,
                     unsigned short* __restrict__ kws,
                     unsigned short* __restrict__ vws)
{
  __shared__ unsigned short lds_a[128 * 32];  // [m][k] bf16, 8 KB
  __shared__ unsigned short lds_b[128 * 32];  // [n][k] bf16, 8 KB

  const int t     = threadIdx.x;
  const int m0    = blockIdx.x * 128;
  const int ntile = blockIdx.y;            // 0..23
  const int p     = ntile >> 3;            // 0=q, 1=k, 2=v
  const int n0    = (ntile & 7) * 128;

  const float* W            = (p == 0) ? Wq : (p == 1) ? Wk : Wv;
  const float* bias         = (p == 0) ? bq : (p == 1) ? bk : bv;
  unsigned short* dst       = (p == 0) ? qws : (p == 1) ? kws : vws;

  const int lane = t & 31, wave = t >> 5;
  const int half = lane >> 4, l15 = lane & 15;
  const int wm   = wave >> 2, wn = wave & 3;    // 2 x 4 wave grid
  const int base = half ? 8 : 0;                // A-fragment K base

  const v8f zero = {0.f, 0.f, 0.f, 0.f, 0.f, 0.f, 0.f, 0.f};
  v8f acc[4][2];
  #pragma unroll
  for (int m = 0; m < 4; ++m) { acc[m][0] = zero; acc[m][1] = zero; }

  const int srow = t >> 1;            // 0..127 staging row
  const int skc  = (t & 1) * 16;      // k sub-offset 0/16

  const float* arow = hs + (size_t)(m0 + srow) * 1024 + skc;
  const float* brow = W  + (size_t)(n0 + srow) * 1024 + skc;

  for (int k0 = 0; k0 < 1024; k0 += 32) {
    __syncthreads();
    {  // stage A (activations) and B (weights), fp32 -> bf16
      const float4* sa = reinterpret_cast<const float4*>(arow + k0);
      const float4* sb = reinterpret_cast<const float4*>(brow + k0);
      unsigned int* da = reinterpret_cast<unsigned int*>(&lds_a[srow * 32 + skc]);
      unsigned int* db = reinterpret_cast<unsigned int*>(&lds_b[srow * 32 + skc]);
      #pragma unroll
      for (int i = 0; i < 4; ++i) {
        float4 fa = sa[i]; float4 fb = sb[i];
        da[2 * i]     = pack2_bf16(fa.x, fa.y);
        da[2 * i + 1] = pack2_bf16(fa.z, fa.w);
        db[2 * i]     = pack2_bf16(fb.x, fb.y);
        db[2 * i + 1] = pack2_bf16(fb.z, fb.w);
      }
      if (k0 + 32 < 1024) {  // prefetch next k-slab into cache (global_prefetch_b8)
        __builtin_prefetch(arow + k0 + 32, 0, 3);
        __builtin_prefetch(brow + k0 + 32, 0, 3);
      }
    }
    __syncthreads();

    Frag16 bf[2];
    #pragma unroll
    for (int n = 0; n < 2; ++n) {
      // B 32x16: col=l15; elems 0..7 = K base..base+7, 8..15 = K base+16..base+23
      const uint4* pb = reinterpret_cast<const uint4*>(&lds_b[(wn * 32 + n * 16 + l15) * 32 + base]);
      bf[n].q[0] = pb[0]; bf[n].q[1] = pb[2];
    }
    #pragma unroll
    for (int m = 0; m < 4; ++m) {
      // A 16x32: lane row = l15; elems 0..7 = K base..base+7, 8..15 = K base+16..base+23
      Frag16 af;
      const uint4* pa = reinterpret_cast<const uint4*>(&lds_a[(wm * 64 + m * 16 + l15) * 32 + base]);
      af.q[0] = pa[0]; af.q[1] = pa[2];
      acc[m][0] = WMMA_BF16(af.v, bf[0].v, acc[m][0]);
      acc[m][1] = WMMA_BF16(af.v, bf[1].v, acc[m][1]);
    }
  }

  // Epilogue: + bias, convert to bf16, scatter into [B, nh, L, 64].
  float bval[2];
  #pragma unroll
  for (int n = 0; n < 2; ++n) bval[n] = bias[n0 + wn * 32 + n * 16 + l15];

  #pragma unroll
  for (int m = 0; m < 4; ++m) {
    #pragma unroll
    for (int n = 0; n < 2; ++n) {
      #pragma unroll
      for (int r = 0; r < 8; ++r) {
        const int gm = m0 + wm * 64 + m * 16 + r + half * 8;   // global row (b*L + l)
        const int gn = n0 + wn * 32 + n * 16 + l15;            // global out-feature
        const float v = acc[m][n][r] + bval[n];
        const int bb = gm >> 10, ll = gm & 1023;
        const int hh = gn >> 6,  dd = gn & 63;
        dst[(((size_t)(bb * 16 + hh) * 1024) + ll) * 64 + dd] = f32_to_bf16_rne(v);
      }
    }
  }
}

// ---------------------------------------------------------------------------
// Kernel 2: flash attention. One block = one (b, h, 64-query tile).
// 128 threads = 4 waves; each wave owns 16 query rows.
// K tile staged with GLOBAL_LOAD_ASYNC_TO_LDS_B128 (ASYNCcnt-tracked DMA).
// ---------------------------------------------------------------------------
__global__ __launch_bounds__(128)
void attn_kernel(const unsigned short* __restrict__ qws,
                 const unsigned short* __restrict__ kws,
                 const unsigned short* __restrict__ vws,
                 float* __restrict__ out)
{
  __shared__ unsigned short k_lds[64 * 64];       // [key][d]  8 KB
  __shared__ unsigned short vt_lds[64 * 64];      // [d][key]  8 KB (transposed V)
  __shared__ unsigned short p_lds[4][16 * 64];    // per-wave P tile, 8 KB

  const int t    = threadIdx.x;
  const int lane = t & 31, wave = t >> 5;
  const int half = lane >> 4, l15 = lane & 15;
  const int base = half ? 8 : 0;
  const int qt = blockIdx.x, h = blockIdx.y, b = blockIdx.z;
  const float scale = 0.125f;   // 1/sqrt(64)

  const size_t head_off = (size_t)(b * 16 + h) * 1024 * 64;
  const unsigned short* Qh = qws + head_off;
  const unsigned short* Kh = kws + head_off;
  const unsigned short* Vh = vws + head_off;

  const int q0 = qt * 64 + wave * 16;

  // Q fragments (A-operand, 2 chunks over d) held in registers.
  Frag16 qf[2];
  #pragma unroll
  for (int c = 0; c < 2; ++c) {
    const uint4* pq = reinterpret_cast<const uint4*>(Qh + (size_t)(q0 + l15) * 64 + c * 32 + base);
    qf[c].q[0] = pq[0]; qf[c].q[1] = pq[2];
  }

  const v8f zero = {0.f, 0.f, 0.f, 0.f, 0.f, 0.f, 0.f, 0.f};
  v8f o[4];
  float mrow[8], lrow[8];
  #pragma unroll
  for (int j = 0; j < 4; ++j) o[j] = zero;
  #pragma unroll
  for (int r = 0; r < 8; ++r) { mrow[r] = -1e30f; lrow[r] = 0.f; }

  const int srow = t >> 1;          // 0..63 staging row (key index within block)
  const int scc  = (t & 1) * 32;    // d sub-offset 0/32

  for (int kb = 0; kb < 16; ++kb) {
    __syncthreads();

    // ---- stage K block [64 key][64 d] directly into LDS (async DMA) ----
    {
      const unsigned short* kg = Kh + (size_t)(kb * 64 + srow) * 64 + scc;
#if defined(__gfx1250__)
      unsigned int kl = (unsigned int)(uintptr_t)&k_lds[srow * 64 + scc];
      unsigned long long kga = (unsigned long long)(uintptr_t)kg;
      asm volatile(
          "global_load_async_to_lds_b128 %0, %1, off\n\t"
          "global_load_async_to_lds_b128 %0, %1, off offset:16\n\t"
          "global_load_async_to_lds_b128 %0, %1, off offset:32\n\t"
          "global_load_async_to_lds_b128 %0, %1, off offset:48"
          :: "v"(kl), "v"(kga) : "memory");
#else
      const uint4* sk = reinterpret_cast<const uint4*>(kg);
      uint4* dk = reinterpret_cast<uint4*>(&k_lds[srow * 64 + scc]);
      dk[0] = sk[0]; dk[1] = sk[1]; dk[2] = sk[2]; dk[3] = sk[3];
#endif
    }

    // ---- stage V block transposed: vt[d][key] (b128 loads, b16 scatter) ----
    {
      const uint4* sv = reinterpret_cast<const uint4*>(Vh + (size_t)(kb * 64 + srow) * 64 + scc);
      uint4 vr[4];
      #pragma unroll
      for (int i = 0; i < 4; ++i) vr[i] = sv[i];
      #pragma unroll
      for (int i = 0; i < 4; ++i) {
        const unsigned int w0 = vr[i].x, w1 = vr[i].y, w2 = vr[i].z, w3 = vr[i].w;
        const int dbase = scc + i * 8;
        vt_lds[(dbase + 0) * 64 + srow] = (unsigned short)(w0);
        vt_lds[(dbase + 1) * 64 + srow] = (unsigned short)(w0 >> 16);
        vt_lds[(dbase + 2) * 64 + srow] = (unsigned short)(w1);
        vt_lds[(dbase + 3) * 64 + srow] = (unsigned short)(w1 >> 16);
        vt_lds[(dbase + 4) * 64 + srow] = (unsigned short)(w2);
        vt_lds[(dbase + 5) * 64 + srow] = (unsigned short)(w2 >> 16);
        vt_lds[(dbase + 6) * 64 + srow] = (unsigned short)(w3);
        vt_lds[(dbase + 7) * 64 + srow] = (unsigned short)(w3 >> 16);
      }
    }

#if defined(__gfx1250__)
    asm volatile("s_wait_asynccnt 0" ::: "memory");   // drain async K DMA
#endif
    __syncthreads();

    // S = Q @ K^T : 4 key sub-tiles x 2 d-chunks
    v8f s4[4];
    #pragma unroll
    for (int j = 0; j < 4; ++j) s4[j] = zero;
    #pragma unroll
    for (int c = 0; c < 2; ++c) {
      #pragma unroll
      for (int j = 0; j < 4; ++j) {
        Frag16 kf;  // B-operand: col = key (j*16+l15), K = d chunk
        const uint4* pk = reinterpret_cast<const uint4*>(&k_lds[(j * 16 + l15) * 64 + c * 32 + base]);
        kf.q[0] = pk[0]; kf.q[1] = pk[2];
        s4[j] = WMMA_BF16(qf[c].v, kf.v, s4[j]);
      }
    }

    // Online softmax (row = r + 8*half, cols striped over 16 lanes of the half).
    #pragma unroll
    for (int r = 0; r < 8; ++r) {
      float mx = fmaxf(fmaxf(s4[0][r], s4[1][r]), fmaxf(s4[2][r], s4[3][r]));
      #pragma unroll
      for (int off = 1; off < 16; off <<= 1) mx = fmaxf(mx, __shfl_xor(mx, off, 32));
      const float mblk = mx * scale;
      const float mnew = fmaxf(mrow[r], mblk);
      const float alpha = __expf(mrow[r] - mnew);
      mrow[r] = mnew;
      float sum = 0.f;
      #pragma unroll
      for (int j = 0; j < 4; ++j) {
        const float pv = __expf(s4[j][r] * scale - mnew);
        sum += pv;
        p_lds[wave][(r + half * 8) * 64 + j * 16 + l15] = f32_to_bf16_rne(pv);
      }
      #pragma unroll
      for (int off = 1; off < 16; off <<= 1) sum += __shfl_xor(sum, off, 32);
      lrow[r] = lrow[r] * alpha + sum;
      #pragma unroll
      for (int j = 0; j < 4; ++j) o[j][r] *= alpha;
    }

    // Same-wave LDS RAW: wait for P stores before fragment reads.
    asm volatile("s_wait_dscnt 0" ::: "memory");

    // O += P @ V : P as A-operand from per-wave LDS, V^T as B-operand.
    #pragma unroll
    for (int c = 0; c < 2; ++c) {
      Frag16 pf;
      const uint4* pp = reinterpret_cast<const uint4*>(&p_lds[wave][l15 * 64 + c * 32 + base]);
      pf.q[0] = pp[0]; pf.q[1] = pp[2];
      #pragma unroll
      for (int j = 0; j < 4; ++j) {
        Frag16 vf;  // col = d (j*16+l15), K = key chunk -> contiguous row of vt_lds
        const uint4* pv = reinterpret_cast<const uint4*>(&vt_lds[(j * 16 + l15) * 64 + c * 32 + base]);
        vf.q[0] = pv[0]; vf.q[1] = pv[2];
        o[j] = WMMA_BF16(pf.v, vf.v, o[j]);
      }
    }
  }

  // Normalize and store ctx fp32: out[b, l, h*64 + d].
  #pragma unroll
  for (int r = 0; r < 8; ++r) {
    const float inv = (lrow[r] > 0.f) ? 1.0f / lrow[r] : 0.f;
    const int gq = q0 + r + half * 8;
    #pragma unroll
    for (int j = 0; j < 4; ++j) {
      const int d = j * 16 + l15;
      out[((size_t)(b * 1024 + gq)) * 1024 + h * 64 + d] = o[j][r] * inv;
    }
  }
}

// ---------------------------------------------------------------------------
extern "C" void kernel_launch(void* const* d_in, const int* in_sizes, int n_in,
                              void* d_out, int out_size, void* d_ws, size_t ws_size,
                              hipStream_t stream) {
  const float* hs = (const float*)d_in[0];
  const float* Wq = (const float*)d_in[1];
  const float* bq = (const float*)d_in[2];
  const float* Wk = (const float*)d_in[3];
  const float* bk = (const float*)d_in[4];
  const float* Wv = (const float*)d_in[5];
  const float* bv = (const float*)d_in[6];
  float* out = (float*)d_out;

  // Workspace: bf16 Q, K, V in [B, nh, L, 64] layout (8 Mi elements each).
  unsigned short* qws = (unsigned short*)d_ws;
  unsigned short* kws = qws + (size_t)8192 * 1024;
  unsigned short* vws = kws + (size_t)8192 * 1024;

  qkv_proj_kernel<<<dim3(64, 24), 256, 0, stream>>>(hs, Wq, bq, Wk, bk, Wv, bv,
                                                    qws, kws, vws);
  attn_kernel<<<dim3(16, 16, 8), 128, 0, stream>>>(qws, kws, vws, out);
}
